// Qwen3QuantizedMLP_73761768341751
// MI455X (gfx1250) — compile-verified, matrix operand order checked by reading, not run on hardware
//
#include <hip/hip_runtime.h>

// Problem sizes (B=1 folded into S): x[64,2048], gate/up[2048,6144], down[6144,2048]
#define S_TOK 64
#define HID   2048
#define IDIM  6144
#define NWAVE 8          // waves per block (K-split factor)

typedef _Float16 v16h __attribute__((ext_vector_type(16)));
typedef float    v8f  __attribute__((ext_vector_type(8)));

union HF16 { v16h v; _Float16 h[16]; };

// Reference: w.astype(float8_e4m3fn).astype(f32). Hardware f32->fp8 RNE
// converter, pairwise (one v_cvt_pk_fp8_f32 per TWO weights), then back to
// f32 and packed to f16 (fp8 values are exact in f16).
__device__ __forceinline__ void quant2_e4m3(float w0, float w1,
                                            _Float16& h0, _Float16& h1) {
  int pk = __builtin_amdgcn_cvt_pk_fp8_f32(w0, w1, 0, false);
  h0 = (_Float16)__builtin_amdgcn_cvt_f32_fp8(pk, 0);
  h1 = (_Float16)__builtin_amdgcn_cvt_f32_fp8(pk, 1);
}

// ---------------- Kernel 1: gate/up GEMMs + SwiGLU -> f16 fuse ----------------
// One block per 16-column tile of I (384 blocks). 8 waves split K=2048 into
// 256-deep slices; f32 partials are reduced through LDS, then the SwiGLU
// epilogue writes the fused activation.
__global__ void __launch_bounds__(256)
qwen3_mlp_gate_up(const float* __restrict__ x,      // [64, 2048]
                  const float* __restrict__ gw,     // [2048, 6144]
                  const float* __restrict__ uw,     // [2048, 6144]
                  const float* __restrict__ gscale, // [6144]
                  const float* __restrict__ uscale, // [6144]
                  _Float16* __restrict__ fuse)      // [64, 6144] f16
{
  __shared__ float red[NWAVE * 2048];   // 64 KB: per wave, 1024 gate + 1024 up

  const int lane  = threadIdx.x & 31;
  const int wave  = threadIdx.x >> 5;
  const int nt    = blockIdx.x;              // 0..383
  const int lmod  = lane & 15;
  const int lhalf = lane >> 4;
  const int koff  = lhalf * 8;               // K sub-block per half-wave
  const int col   = nt * 16 + lmod;
  const int kbeg  = wave * (HID / NWAVE);    // 256-deep K slice per wave

  v8f accg[4], accu[4];
#pragma unroll
  for (int mt = 0; mt < 4; ++mt) { accg[mt] = {}; accu[mt] = {}; }

  for (int k0 = kbeg; k0 < kbeg + HID / NWAVE; k0 += 32) {
    HF16 bg, bu;
#pragma unroll
    for (int e = 0; e < 8; e += 2) {
      const long k = k0 + koff + e;
      quant2_e4m3(gw[k * IDIM + col],        gw[(k + 1) * IDIM + col],
                  bg.h[e], bg.h[e + 1]);
      quant2_e4m3(gw[(k + 16) * IDIM + col], gw[(k + 17) * IDIM + col],
                  bg.h[e + 8], bg.h[e + 9]);
      quant2_e4m3(uw[k * IDIM + col],        uw[(k + 1) * IDIM + col],
                  bu.h[e], bu.h[e + 1]);
      quant2_e4m3(uw[(k + 16) * IDIM + col], uw[(k + 17) * IDIM + col],
                  bu.h[e + 8], bu.h[e + 9]);
    }
#pragma unroll
    for (int mt = 0; mt < 4; ++mt) {
      HF16 a;
      const int row = mt * 16 + lmod;
#pragma unroll
      for (int e = 0; e < 8; ++e) {
        const int k = k0 + koff + e;
        a.h[e]     = (_Float16)x[row * HID + k];
        a.h[e + 8] = (_Float16)x[row * HID + k + 16];
      }
      accg[mt] = __builtin_amdgcn_wmma_f32_16x16x32_f16(
          false, a.v, false, bg.v, (short)0, accg[mt], false, false);
      accu[mt] = __builtin_amdgcn_wmma_f32_16x16x32_f16(
          false, a.v, false, bu.v, (short)0, accu[mt], false, false);
    }
  }

  // Park partials: slot = mt*256 + r*32 + lane, per-wave region of 2048 f32.
#pragma unroll
  for (int mt = 0; mt < 4; ++mt)
#pragma unroll
    for (int r = 0; r < 8; ++r) {
      red[wave * 2048 +        mt * 256 + r * 32 + lane] = accg[mt][r];
      red[wave * 2048 + 1024 + mt * 256 + r * 32 + lane] = accu[mt][r];
    }
  __syncthreads();

  // Cross-wave reduction + SwiGLU epilogue: 1024 output slots / 256 threads.
#pragma unroll
  for (int i = 0; i < 4; ++i) {
    const int slot = threadIdx.x + i * 256;
    float g = 0.f, u = 0.f;
#pragma unroll
    for (int w = 0; w < NWAVE; ++w) {
      g += red[w * 2048 + slot];
      u += red[w * 2048 + 1024 + slot];
    }
    const int mt = slot >> 8;
    const int r  = (slot >> 5) & 7;
    const int ln = slot & 31;
    const int row = mt * 16 + r + ((ln >> 4) << 3);   // C/D VGPR layout
    const int c   = nt * 16 + (ln & 15);
    g *= gscale[c];
    u *= uscale[c];
    const float s = 1.0f / (1.0f + __expf(-g));
    fuse[(long)row * IDIM + c] = (_Float16)(g * s * u);
  }
}

// ---------------- Kernel 2: down projection ----------------
// One block per 16-column tile of H (128 blocks); 8 waves split K=6144 into
// 768-deep slices, LDS reduction, scaled f32 output.
__global__ void __launch_bounds__(256)
qwen3_mlp_down(const _Float16* __restrict__ fuse,   // [64, 6144] f16
               const float* __restrict__ dw,        // [6144, 2048]
               const float* __restrict__ dscale,    // [2048]
               float* __restrict__ out)             // [64, 2048]
{
  __shared__ float red[NWAVE * 1024];   // 32 KB

  const int lane  = threadIdx.x & 31;
  const int wave  = threadIdx.x >> 5;
  const int nt    = blockIdx.x;              // 0..127
  const int lmod  = lane & 15;
  const int lhalf = lane >> 4;
  const int koff  = lhalf * 8;
  const int col   = nt * 16 + lmod;
  const int kbeg  = wave * (IDIM / NWAVE);   // 768-deep K slice

  v8f acc[4];
#pragma unroll
  for (int mt = 0; mt < 4; ++mt) acc[mt] = {};

  for (int k0 = kbeg; k0 < kbeg + IDIM / NWAVE; k0 += 32) {
    HF16 b;
#pragma unroll
    for (int e = 0; e < 8; e += 2) {
      const long k = k0 + koff + e;
      quant2_e4m3(dw[k * HID + col],        dw[(k + 1) * HID + col],
                  b.h[e], b.h[e + 1]);
      quant2_e4m3(dw[(k + 16) * HID + col], dw[(k + 17) * HID + col],
                  b.h[e + 8], b.h[e + 9]);
    }
#pragma unroll
    for (int mt = 0; mt < 4; ++mt) {
      HF16 a;
      const int row = mt * 16 + lmod;
#pragma unroll
      for (int e = 0; e < 8; ++e) {
        const int k = k0 + koff + e;
        a.h[e]     = fuse[(long)row * IDIM + k];
        a.h[e + 8] = fuse[(long)row * IDIM + k + 16];
      }
      acc[mt] = __builtin_amdgcn_wmma_f32_16x16x32_f16(
          false, a.v, false, b.v, (short)0, acc[mt], false, false);
    }
  }

#pragma unroll
  for (int mt = 0; mt < 4; ++mt)
#pragma unroll
    for (int r = 0; r < 8; ++r)
      red[wave * 1024 + mt * 256 + r * 32 + lane] = acc[mt][r];
  __syncthreads();

#pragma unroll
  for (int i = 0; i < 4; ++i) {
    const int slot = threadIdx.x + i * 256;
    float v = 0.f;
#pragma unroll
    for (int w = 0; w < NWAVE; ++w) v += red[w * 1024 + slot];
    const int mt = slot >> 8;
    const int r  = (slot >> 5) & 7;
    const int ln = slot & 31;
    const int row = mt * 16 + r + ((ln >> 4) << 3);
    const int c   = nt * 16 + (ln & 15);
    out[(long)row * HID + c] = v * dscale[c];
  }
}

extern "C" void kernel_launch(void* const* d_in, const int* in_sizes, int n_in,
                              void* d_out, int out_size, void* d_ws, size_t ws_size,
                              hipStream_t stream) {
  const float* x   = (const float*)d_in[0];
  const float* gw  = (const float*)d_in[1];
  const float* uw  = (const float*)d_in[2];
  const float* dw  = (const float*)d_in[3];
  const float* gsc = (const float*)d_in[4];
  const float* usc = (const float*)d_in[5];
  const float* dsc = (const float*)d_in[6];
  float* out = (float*)d_out;
  _Float16* fuse = (_Float16*)d_ws;   // 64*6144*2 = 786 KB scratch

  // 384 blocks x 8 waves (K-split) = 3072 waves streaming the gate/up weights.
  qwen3_mlp_gate_up<<<IDIM / 16, 256, 0, stream>>>(x, gw, uw, gsc, usc, fuse);
  // 128 blocks x 8 waves for the down projection.
  qwen3_mlp_down<<<HID / 16, 256, 0, stream>>>(fuse, dw, dsc, out);
}
